// LinearGatedMHSA2d_58463094833343
// MI455X (gfx1250) — matrix operand order repaired and checked by reading, not compiled, not run on hardware
//
#include <hip/hip_runtime.h>
#include <hip/hip_bf16.h>
#include <math.h>

// Problem constants (fixed by the reference)
#define CDIM   256
#define NHEAD  8
#define DHEAD  32
#define NTOK   16384               // H*W = 128*128
#define NBATCH 8
#define TTOT   (NBATCH * NTOK)     // 131072 tokens
#define QKV3   (3 * CDIM)          // 768

typedef __attribute__((ext_vector_type(16))) __bf16 v16bf;
typedef __attribute__((ext_vector_type(8)))  __bf16 v8bf;
typedef __attribute__((ext_vector_type(8)))  float  v8f;

// ---------- bf16 helpers (raw ushort storage, RNE convert) ----------
static __device__ __forceinline__ unsigned short f2bf(float f) {
  unsigned u = __float_as_uint(f);
  unsigned r = u + 0x7FFFu + ((u >> 16) & 1u);
  return (unsigned short)(r >> 16);
}
static __device__ __forceinline__ float bf2f(unsigned short s) {
  return __uint_as_float(((unsigned)s) << 16);
}

// ---------- WMMA operand fragment loaders (wave32, 16x16x32 bf16) ----------
union BF8 { uint4 u; v8bf h; };

static __device__ __forceinline__ v16bf make16(v8bf lo, v8bf hi) {
  v16bf a;
#pragma unroll
  for (int i = 0; i < 8; ++i) { a[i] = lo[i]; a[8 + i] = hi[i]; }
  return a;
}
// A operand: row = lane%16 of row-major A tile; lanes 0-15 hold K {kk+0..7, kk+16..23},
// lanes 16-31 hold K {kk+8..15, kk+24..31}.  half = (lane>=16). Works for global or LDS rows.
static __device__ __forceinline__ v16bf load_a_frag(const unsigned short* rowp, int kk, int half) {
  BF8 lo, hi;
  lo.u = *(const uint4*)(rowp + kk + half * 8);
  hi.u = *(const uint4*)(rowp + kk + half * 8 + 16);
  return make16(lo.h, hi.h);
}
// B operand from row-major B^T: col = lane%16; lanes 0-15 hold K {kk+0..7, kk+16..23},
// lanes 16-31 hold K {kk+8..15, kk+24..31} (same order as A).
static __device__ __forceinline__ v16bf load_b_frag(const unsigned short* rowp, int kk, int half) {
  const unsigned short* p = rowp + kk + half * 8;
  BF8 lo, hi;
  lo.u = *(const uint4*)(p);
  hi.u = *(const uint4*)(p + 16);
  return make16(lo.h, hi.h);
}
static __device__ __forceinline__ v8f wmma_bf16(v16bf a, v16bf b, v8f c) {
  return __builtin_amdgcn_wmma_f32_16x16x32_bf16(false, a, false, b, (short)0, c, false, false);
}

// ---------- Kernel 1: transpose + bf16-cast weights ----------
__global__ void cvt_weights_kernel(const float* __restrict__ wqkv, const float* __restrict__ wproj,
                                   unsigned short* __restrict__ wqkvT, unsigned short* __restrict__ wprojT) {
  int i = blockIdx.x * 256 + threadIdx.x;
  if (i < QKV3 * CDIM) {
    int o = i / CDIM, c = i % CDIM;
    wqkvT[o * CDIM + c] = f2bf(wqkv[c * QKV3 + o]);
  }
  if (i < CDIM * CDIM) {
    int o = i / CDIM, c = i % CDIM;
    wprojT[o * CDIM + c] = f2bf(wproj[c * CDIM + o]);
  }
}

// ---------- Kernel 2: LayerNorm over C, token-major bf16 output ----------
// x layout [B,C,N]; one thread per token; reads coalesced along n (x is L2-resident: 128MB < 192MB L2).
__global__ void ln_kernel(const float* __restrict__ x, const float* __restrict__ gamma,
                          const float* __restrict__ beta, unsigned short* __restrict__ xt) {
  int t = blockIdx.x * 256 + threadIdx.x;          // global token
  int b = t >> 14, n = t & (NTOK - 1);
  const float* base = x + (size_t)b * CDIM * NTOK + n;
  float s = 0.f, s2 = 0.f;
#pragma unroll 4
  for (int c = 0; c < CDIM; ++c) { float v = base[c * NTOK]; s += v; s2 += v * v; }
  float mu = s * (1.0f / CDIM);
  float var = s2 * (1.0f / CDIM) - mu * mu;
  float rstd = rsqrtf(var + 1e-5f);
  unsigned short* dst = xt + (size_t)t * CDIM;
#pragma unroll 4
  for (int c = 0; c < CDIM; ++c) {
    float v = (base[c * NTOK] - mu) * rstd * gamma[c] + beta[c];
    dst[c] = f2bf(v);
  }
}

// ---------- Kernel 3: QKV GEMM (WMMA) with fused phi, transposed K/V stores ----------
// grid (2048, 6), block 256 (8 waves). Block covers 64 tokens staged once in LDS;
// each wave owns one 16-wide out-channel tile and register-blocks 4 m-tiles (4x B reuse).
__global__ void qkv_kernel(const unsigned short* __restrict__ xt, const unsigned short* __restrict__ wqkvT,
                           unsigned short* __restrict__ qphi, unsigned short* __restrict__ kphiT,
                           unsigned short* __restrict__ vT) {
  __shared__ __align__(16) unsigned short lds_a[64 * CDIM];   // 32 KiB
  int tid = threadIdx.x;
  int lane = tid & 31, w = tid >> 5;
  int tblock = blockIdx.x * 64;
  {                                                  // 64 contiguous token rows: linear 32KB copy
    const uint4* src = (const uint4*)(xt + (size_t)tblock * CDIM);
    uint4* dst = (uint4*)lds_a;
#pragma unroll
    for (int i = 0; i < 8; ++i) dst[tid + i * 256] = src[tid + i * 256];
  }
  __syncthreads();

  int ntile = blockIdx.y * 8 + w;                    // 0..47
  int obase = ntile * 16;
  int half = (lane >= 16) ? 1 : 0;
  const unsigned short* brow = wqkvT + (size_t)(obase + (lane & 15)) * CDIM;
  const unsigned short* a0 = lds_a + (lane & 15) * CDIM;
  v8f acc[4] = {};
#pragma unroll
  for (int kk = 0; kk < CDIM; kk += 32) {
    v16bf b = load_b_frag(brow, kk, half);
#pragma unroll
    for (int mt = 0; mt < 4; ++mt) {
      v16bf a = load_a_frag(a0 + mt * 16 * CDIM, kk, half);
      acc[mt] = wmma_bf16(a, b, acc[mt]);
    }
  }

  int col = obase + (lane & 15);
  int rowoff = (lane >= 16) ? 8 : 0;
#pragma unroll
  for (int mt = 0; mt < 4; ++mt) {
    int tbase = tblock + mt * 16;
#pragma unroll
    for (int r = 0; r < 8; ++r) {
      int t = tbase + r + rowoff;
      int b = t >> 14, n = t & (NTOK - 1);
      float val = acc[mt][r];
      if (col < CDIM) {                              // Q -> phi, token-major [bh][n][d]
        int head = col >> 5, dd = col & 31;
        float p = val > 0.f ? val + 1.f : __expf(val);
        qphi[(((size_t)(b * NHEAD + head) * NTOK + n) << 5) + dd] = f2bf(p);
      } else if (col < 2 * CDIM) {                   // K -> phi, d-major [bh][d][n]
        int cc = col - CDIM; int head = cc >> 5, dd = cc & 31;
        float p = val > 0.f ? val + 1.f : __expf(val);
        kphiT[(((size_t)(b * NHEAD + head) * DHEAD + dd) << 14) + n] = f2bf(p);
      } else {                                       // V -> d-major [bh][d][n]
        int cc = col - 2 * CDIM; int head = cc >> 5, dd = cc & 31;
        vT[(((size_t)(b * NHEAD + head) * DHEAD + dd) << 14) + n] = f2bf(val);
      }
    }
  }
}

// ---------- Kernel 4: kv = k_phi^T @ v  (32x32 per (b,h); K=16384 split over 8 waves) ----------
__global__ void kv_kernel(const unsigned short* __restrict__ kphiT, const unsigned short* __restrict__ vT,
                          unsigned short* __restrict__ kvT) {
  __shared__ float red[8][4][256];                  // 32 KB
  int bh = blockIdx.x;
  int lane = threadIdx.x & 31, w = threadIdx.x >> 5;
  int half = (lane >= 16) ? 1 : 0;
  const unsigned short* kb = kphiT + (size_t)bh * DHEAD * NTOK;
  const unsigned short* vb = vT + (size_t)bh * DHEAD * NTOK;
  const unsigned short* arow0 = kb + (size_t)(0 + (lane & 15)) * NTOK;
  const unsigned short* arow1 = kb + (size_t)(16 + (lane & 15)) * NTOK;
  const unsigned short* brow0 = vb + (size_t)(0 + (lane & 15)) * NTOK;
  const unsigned short* brow1 = vb + (size_t)(16 + (lane & 15)) * NTOK;
  v8f acc0 = {}, acc1 = {}, acc2 = {}, acc3 = {};
  int k0 = w * (NTOK / 8);
  for (int kk = k0; kk < k0 + NTOK / 8; kk += 32) {
    v16bf a0 = load_a_frag(arow0, kk, half);
    v16bf a1 = load_a_frag(arow1, kk, half);
    v16bf b0 = load_b_frag(brow0, kk, half);
    v16bf b1 = load_b_frag(brow1, kk, half);
    acc0 = wmma_bf16(a0, b0, acc0);
    acc1 = wmma_bf16(a0, b1, acc1);
    acc2 = wmma_bf16(a1, b0, acc2);
    acc3 = wmma_bf16(a1, b1, acc3);
  }
#pragma unroll
  for (int r = 0; r < 8; ++r) {
    red[w][0][r * 32 + lane] = acc0[r];
    red[w][1][r * 32 + lane] = acc1[r];
    red[w][2][r * 32 + lane] = acc2[r];
    red[w][3][r * 32 + lane] = acc3[r];
  }
  __syncthreads();
  if (w < 4) {
    int tile = w;
    int mt = tile >> 1, nt = tile & 1;
    int e = nt * 16 + (lane & 15);
    int rowoff = (lane >= 16) ? 8 : 0;
#pragma unroll
    for (int r = 0; r < 8; ++r) {
      float s = 0.f;
#pragma unroll
      for (int ww = 0; ww < 8; ++ww) s += red[ww][tile][r * 32 + lane];
      int m = mt * 16 + r + rowoff;
      kvT[bh * 1024 + e * 32 + m] = f2bf(s);        // store kv transposed: [e][m]
    }
  }
}

// ---------- Kernel 5: k_sum[bh][d] = sum_n k_phi ----------
__global__ void ksum_kernel(const unsigned short* __restrict__ kphiT, float* __restrict__ ksum) {
  int lane = threadIdx.x & 31, w = threadIdx.x >> 5;
  int row = blockIdx.x * 8 + w;                     // 0..2047
  const unsigned short* p = kphiT + (size_t)row * NTOK;
  float s = 0.f;
  for (int i = lane; i < NTOK; i += 32) s += bf2f(p[i]);
#pragma unroll
  for (int off = 16; off >= 1; off >>= 1) s += __shfl_xor(s, off, 32);
  if (lane == 0) ksum[row] = s;
}

// ---------- Kernel 6: zg[bh][n] = (1/(q_phi . k_sum + eps)) * sigmoid(xt.w_gate + b_gate) ----------
__global__ void zg_kernel(const unsigned short* __restrict__ qphi, const float* __restrict__ ksum,
                          const unsigned short* __restrict__ xt, const float* __restrict__ wgate,
                          const float* __restrict__ bgate, float* __restrict__ zg) {
  int idx = blockIdx.x * 256 + threadIdx.x;         // < 1048576
  int n = idx & (NTOK - 1);
  int bh = idx >> 14;
  int b = bh >> 3, head = bh & 7;
  const unsigned short* qr = qphi + (((size_t)bh * NTOK + n) << 5);
  float den = 0.f;
#pragma unroll
  for (int dd = 0; dd < DHEAD; ++dd) den += bf2f(qr[dd]) * ksum[bh * 32 + dd];
  float z = 1.f / (den + 1e-6f);
  const unsigned short* xr = xt + (size_t)(b * NTOK + n) * CDIM;
  float g = bgate[head];
#pragma unroll 4
  for (int c = 0; c < CDIM; ++c) g += bf2f(xr[c]) * wgate[c * NHEAD + head];
  float gate = 1.f / (1.f + __expf(-g));
  zg[idx] = z * gate;
}

// ---------- Kernel 7: out = (q_phi @ kv) * zg, merge heads (bf16, token-major) ----------
__global__ void attn_out_kernel(const unsigned short* __restrict__ qphi, const unsigned short* __restrict__ kvT,
                                const float* __restrict__ zg, unsigned short* __restrict__ merged) {
  int lane = threadIdx.x & 31, w = threadIdx.x >> 5;
  int tileid = blockIdx.x * 8 + w;                  // < 65536
  int bh = tileid >> 10;
  int mt = tileid & 1023;
  int b = bh >> 3, head = bh & 7;
  int tbase = mt * 16;
  int half = (lane >= 16) ? 1 : 0;
  const unsigned short* arow = qphi + (((size_t)bh * NTOK + tbase + (lane & 15)) << 5);
  v16bf a = load_a_frag(arow, 0, half);             // K = d = 32: one k-step
  const unsigned short* kvb = kvT + bh * 1024;
  v16bf b0 = load_b_frag(kvb + (0 + (lane & 15)) * 32, 0, half);
  v16bf b1 = load_b_frag(kvb + (16 + (lane & 15)) * 32, 0, half);
  v8f acc0 = {}, acc1 = {};
  acc0 = wmma_bf16(a, b0, acc0);
  acc1 = wmma_bf16(a, b1, acc1);
  int rowoff = (lane >= 16) ? 8 : 0;
  int colb = lane & 15;
#pragma unroll
  for (int r = 0; r < 8; ++r) {
    int n = tbase + r + rowoff;
    float zv = zg[bh * NTOK + n];
    size_t tg = (size_t)(b * NTOK + n) * CDIM + head * DHEAD;
    merged[tg + colb] = f2bf(acc0[r] * zv);
    merged[tg + 16 + colb] = f2bf(acc1[r] * zv);
  }
}

// ---------- Kernel 8: proj GEMM + residual, output in [B,C,H,W] ----------
// grid (2048, 2), block 256: 64-token A tile staged in LDS, 4 m-tiles per wave.
__global__ void proj_kernel(const unsigned short* __restrict__ merged, const unsigned short* __restrict__ wprojT,
                            const float* __restrict__ x, float* __restrict__ out) {
  __shared__ __align__(16) unsigned short lds_a[64 * CDIM];   // 32 KiB
  int tid = threadIdx.x;
  int lane = tid & 31, w = tid >> 5;
  int tblock = blockIdx.x * 64;
  {
    const uint4* src = (const uint4*)(merged + (size_t)tblock * CDIM);
    uint4* dst = (uint4*)lds_a;
#pragma unroll
    for (int i = 0; i < 8; ++i) dst[tid + i * 256] = src[tid + i * 256];
  }
  __syncthreads();

  int ntile = blockIdx.y * 8 + w;                   // 0..15
  int cbase = ntile * 16;
  int half = (lane >= 16) ? 1 : 0;
  const unsigned short* brow = wprojT + (size_t)(cbase + (lane & 15)) * CDIM;
  const unsigned short* a0 = lds_a + (lane & 15) * CDIM;
  v8f acc[4] = {};
#pragma unroll
  for (int kk = 0; kk < CDIM; kk += 32) {
    v16bf b = load_b_frag(brow, kk, half);
#pragma unroll
    for (int mt = 0; mt < 4; ++mt) {
      v16bf a = load_a_frag(a0 + mt * 16 * CDIM, kk, half);
      acc[mt] = wmma_bf16(a, b, acc[mt]);
    }
  }

  int c = cbase + (lane & 15);
  int rowoff = (lane >= 16) ? 8 : 0;
#pragma unroll
  for (int mt = 0; mt < 4; ++mt) {
    int tbase = tblock + mt * 16;
#pragma unroll
    for (int r = 0; r < 8; ++r) {
      int t = tbase + r + rowoff;
      int b = t >> 14, n = t & (NTOK - 1);
      size_t addr = (size_t)(b * CDIM + c) * NTOK + n;
      out[addr] = x[addr] + acc[mt][r];
    }
  }
}

// ---------- Launcher ----------
extern "C" void kernel_launch(void* const* d_in, const int* in_sizes, int n_in,
                              void* d_out, int out_size, void* d_ws, size_t ws_size,
                              hipStream_t stream) {
  (void)in_sizes; (void)n_in; (void)out_size; (void)ws_size;
  const float* x      = (const float*)d_in[0];
  const float* gamma  = (const float*)d_in[1];
  const float* beta   = (const float*)d_in[2];
  const float* wqkv   = (const float*)d_in[3];
  const float* wgate  = (const float*)d_in[4];
  const float* bgate  = (const float*)d_in[5];
  const float* wproj  = (const float*)d_in[6];
  float* out = (float*)d_out;

  char* ws = (char*)d_ws;
  size_t off = 0;
  auto take = [&](size_t bytes) { char* p = ws + off; off += (bytes + 255) & ~(size_t)255; return p; };
  unsigned short* xt     = (unsigned short*)take((size_t)TTOT * CDIM * 2);        // 64 MiB
  unsigned short* wqkvT  = (unsigned short*)take((size_t)QKV3 * CDIM * 2);
  unsigned short* wprojT = (unsigned short*)take((size_t)CDIM * CDIM * 2);
  unsigned short* qphi   = (unsigned short*)take((size_t)TTOT * CDIM * 2);        // 64 MiB
  unsigned short* kphiT  = (unsigned short*)take((size_t)TTOT * CDIM * 2);        // 64 MiB (reused as merged)
  unsigned short* vT     = (unsigned short*)take((size_t)TTOT * CDIM * 2);        // 64 MiB
  unsigned short* kvT    = (unsigned short*)take((size_t)NBATCH * NHEAD * 32 * 32 * 2);
  float*          ksum   = (float*)take((size_t)NBATCH * NHEAD * DHEAD * 4);
  float*          zg     = (float*)take((size_t)NBATCH * NHEAD * NTOK * 4);       // 4 MiB
  unsigned short* merged = kphiT;   // alias: k_phiT dead after ksum_kernel

  cvt_weights_kernel<<<(QKV3 * CDIM + 255) / 256, 256, 0, stream>>>(wqkv, wproj, wqkvT, wprojT);
  ln_kernel<<<TTOT / 256, 256, 0, stream>>>(x, gamma, beta, xt);
  qkv_kernel<<<dim3(TTOT / 64, 6), 256, 0, stream>>>(xt, wqkvT, qphi, kphiT, vT);
  kv_kernel<<<NBATCH * NHEAD, 256, 0, stream>>>(kphiT, vT, kvT);
  ksum_kernel<<<(NBATCH * NHEAD * DHEAD) / 8, 256, 0, stream>>>(kphiT, ksum);
  zg_kernel<<<(NBATCH * NHEAD * NTOK) / 256, 256, 0, stream>>>(qphi, ksum, xt, wgate, bgate, zg);
  attn_out_kernel<<<(NBATCH * NHEAD * (NTOK / 16)) / 8, 256, 0, stream>>>(qphi, kvT, zg, merged);
  proj_kernel<<<dim3(TTOT / 64, 2), 256, 0, stream>>>(merged, wprojT, x, out);
}
